// Pointnet2_based_52656299049077
// MI455X (gfx1250) — matrix operand order repaired and checked by reading, not compiled, not run on hardware
//
#include <hip/hip_runtime.h>
#include <hip/hip_bf16.h>

typedef __attribute__((ext_vector_type(16))) _Float16 v16h;
typedef __attribute__((ext_vector_type(8)))  _Float16 v8h;
typedef __attribute__((ext_vector_type(8)))  float    v8f;

#define BN_EPS_F 1e-5f

// ---------------------------------------------------------------------------
// 1) transpose xyz (B,3,N) -> pts (B,N,3)
// ---------------------------------------------------------------------------
__global__ void pn2_transpose_xyz(const float* __restrict__ x, int B, int N,
                                  float* __restrict__ pts) {
    long i = (long)blockIdx.x * blockDim.x + threadIdx.x;
    long total = (long)B * N;
    if (i >= total) return;
    int b = (int)(i / N);
    int n = (int)(i % N);
    const float* xb = x + (size_t)b * 3 * N;
    float* p = pts + i * 3;
    p[0] = xb[0 * N + n];
    p[1] = xb[1 * N + n];
    p[2] = xb[2 * N + n];
}

// ---------------------------------------------------------------------------
// 2) farthest point sampling: one block per batch (sequential over S picks)
// ---------------------------------------------------------------------------
#define FPS_T 256
__global__ void pn2_fps(const float* __restrict__ pts, int N, int S,
                        int* __restrict__ out_idx) {
    __shared__ float dist[4096];
    __shared__ float rval[FPS_T];
    __shared__ int   ridx[FPS_T];
    __shared__ float cen[3];
    int b = blockIdx.x;
    int t = threadIdx.x;
    const float* P = pts + (size_t)b * N * 3;
    for (int i = t; i < N; i += FPS_T) dist[i] = 1e10f;
    __syncthreads();
    int far = 0;
    for (int s = 0; s < S; ++s) {
        if (t == 0) {
            out_idx[(size_t)b * S + s] = far;
            cen[0] = P[(size_t)far * 3 + 0];
            cen[1] = P[(size_t)far * 3 + 1];
            cen[2] = P[(size_t)far * 3 + 2];
        }
        __syncthreads();
        float best = -1.0f; int bi = N - 1;
        float cx = cen[0], cy = cen[1], cz = cen[2];
        for (int i = t; i < N; i += FPS_T) {
            float dx = P[(size_t)i * 3 + 0] - cx;
            float dy = P[(size_t)i * 3 + 1] - cy;
            float dz = P[(size_t)i * 3 + 2] - cz;
            float d = dx * dx + dy * dy + dz * dz;
            float m = fminf(dist[i], d);
            dist[i] = m;
            if (m > best || (m == best && i < bi)) { best = m; bi = i; }
        }
        rval[t] = best; ridx[t] = bi;
        __syncthreads();
        for (int o = FPS_T / 2; o > 0; o >>= 1) {
            if (t < o) {
                float ov = rval[t + o]; int oi = ridx[t + o];
                if (ov > rval[t] || (ov == rval[t] && oi < ridx[t])) {
                    rval[t] = ov; ridx[t] = oi;
                }
            }
            __syncthreads();
        }
        far = ridx[0];
        // next-iteration __syncthreads() separates this read from rval writes
    }
}

// ---------------------------------------------------------------------------
// 3) gather sampled centers: new_xyz[b][s] = pts[b][fps_idx[b][s]]
// ---------------------------------------------------------------------------
__global__ void pn2_gather_xyz(const float* __restrict__ pts,
                               const int* __restrict__ idx,
                               int B, int N, int S,
                               float* __restrict__ out) {
    int i = blockIdx.x * blockDim.x + threadIdx.x;
    if (i >= B * S) return;
    int b = i / S;
    int j = idx[i];
    const float* p = pts + ((size_t)b * N + j) * 3;
    float* o = out + (size_t)i * 3;
    o[0] = p[0]; o[1] = p[1]; o[2] = p[2];
}

// ---------------------------------------------------------------------------
// 4) ball query: first nsample indices (ascending) with d^2 <= r^2, pad first
// ---------------------------------------------------------------------------
__global__ void pn2_ball_query(const float* __restrict__ pts,
                               const float* __restrict__ new_xyz,
                               int B, int N, int S, int nsample, float r2,
                               int* __restrict__ idx) {
    int i = blockIdx.x * blockDim.x + threadIdx.x;
    if (i >= B * S) return;
    int b = i / S;
    const float* P = pts + (size_t)b * N * 3;
    float qx = new_xyz[(size_t)i * 3 + 0];
    float qy = new_xyz[(size_t)i * 3 + 1];
    float qz = new_xyz[(size_t)i * 3 + 2];
    int* o = idx + (size_t)i * nsample;
    int cnt = 0, first = 0;
    for (int j = 0; j < N && cnt < nsample; ++j) {
        float dx = P[(size_t)j * 3 + 0] - qx;
        float dy = P[(size_t)j * 3 + 1] - qy;
        float dz = P[(size_t)j * 3 + 2] - qz;
        float d = dx * dx + dy * dy + dz * dz;
        if (d <= r2) {
            if (cnt == 0) first = j;
            o[cnt++] = j;
        }
    }
    for (int k = cnt; k < nsample; ++k) o[k] = first;
}

// ---------------------------------------------------------------------------
// 5) build grouped feature rows in f16: row = (b*S+s)*nsample + k
//    cols [0..2] = xyz_j - new_xyz, [3..3+Cp) = points[j], pad to Kpad with 0
// ---------------------------------------------------------------------------
__global__ void pn2_build_feat(const float* __restrict__ pts,
                               const float* __restrict__ new_xyz,
                               const float* __restrict__ points, // may be null
                               const int* __restrict__ idx,
                               int B, int N, int S, int nsample, int Cp, int Kpad,
                               _Float16* __restrict__ feat) {
    long r = (long)blockIdx.x * blockDim.x + threadIdx.x;
    long M = (long)B * S * nsample;
    if (r >= M) return;
    int k = (int)(r % nsample);
    long bs = r / nsample;
    int b = (int)(bs / S);
    int j = idx[bs * nsample + k];
    const float* p = pts + ((size_t)b * N + j) * 3;
    const float* q = new_xyz + (size_t)bs * 3;
    _Float16* f = feat + (size_t)r * Kpad;
    f[0] = (_Float16)(p[0] - q[0]);
    f[1] = (_Float16)(p[1] - q[1]);
    f[2] = (_Float16)(p[2] - q[2]);
    int c = 3;
    if (points) {
        const float* pf = points + ((size_t)b * N + j) * Cp;
        for (int cc = 0; cc < Cp; ++cc) f[c++] = (_Float16)pf[cc];
    }
    for (; c < Kpad; ++c) f[c] = (_Float16)0.0f;
}

// ---------------------------------------------------------------------------
// 6) convert weights f32 (Cout,Cin) -> f16 Wt[Cout][Kpad] (zero-padded K)
// ---------------------------------------------------------------------------
__global__ void pn2_conv_w(const float* __restrict__ w, int Cout, int Cin, int Kpad,
                           _Float16* __restrict__ Wt) {
    int i = blockIdx.x * blockDim.x + threadIdx.x;
    if (i >= Cout * Kpad) return;
    int o = i / Kpad, k = i % Kpad;
    Wt[i] = (k < Cin) ? (_Float16)w[(size_t)o * Cin + k] : (_Float16)0.0f;
}

// folded BN: relu(g*((x+b)*inv)+be) == relu(a*x + c), a=g*inv, c=a*b+be
__global__ void pn2_bn_coef(const float* __restrict__ b, const float* __restrict__ g,
                            const float* __restrict__ be, int C,
                            float* __restrict__ sc, float* __restrict__ sh) {
    int o = blockIdx.x * blockDim.x + threadIdx.x;
    if (o >= C) return;
    float inv = rsqrtf(1.0f + BN_EPS_F);
    float a = g[o] * inv;
    sc[o] = a;
    sh[o] = a * b[o] + be[o];
}

// ---------------------------------------------------------------------------
// 7) WMMA GEMM + fused BN/ReLU.  A: (M x Kpad) f16 row-major.
//    Wt: (Cout x Kpad) f16 row-major (== B^T).  Out: (M x Cout) f16.
//    Grid: (M/128, Cout/64). Block: 256 (8 waves). Wave -> 16x64 strip.
// ---------------------------------------------------------------------------
__device__ inline v16h pn2_join(v8h lo, v8h hi) {
    return __builtin_shufflevector(lo, hi, 0, 1, 2, 3, 4, 5, 6, 7,
                                   8, 9, 10, 11, 12, 13, 14, 15);
}

__device__ inline v16h pn2_load_bfrag(const _Float16* __restrict__ Wt,
                                      int col, int Kpad, int k) {
    const v8h* p = (const v8h*)(Wt + (size_t)col * Kpad + k);
    return pn2_join(p[0], p[2]);
}

__device__ inline void pn2_epilogue(v8f acc, int oc, int rbase, int Cout,
                                    const float* __restrict__ sc,
                                    const float* __restrict__ sh,
                                    _Float16* __restrict__ Out) {
    float a = sc[oc], c = sh[oc];
    _Float16* op = Out + (size_t)rbase * Cout + oc;
#pragma unroll
    for (int v = 0; v < 8; ++v) {
        float x = acc[v] * a + c;
        x = x > 0.0f ? x : 0.0f;
        op[(size_t)v * Cout] = (_Float16)x;
    }
}

__global__ __launch_bounds__(256)
void pn2_gemm_bn_relu(const _Float16* __restrict__ A,
                      const _Float16* __restrict__ Wt,
                      const float* __restrict__ sc,
                      const float* __restrict__ sh,
                      _Float16* __restrict__ Out,
                      int M, int Kpad, int Cout) {
    const int lane = threadIdx.x & 31;
    const int wave = threadIdx.x >> 5;
    const int mBase = blockIdx.x * 128 + wave * 16;
    const int nBase = blockIdx.y * 64;

    v8f acc0 = {}, acc1 = {}, acc2 = {}, acc3 = {};

    // A fragment layout (16x32 f16): lanes 0-15 -> M=lane, K 0-7 & 16-23;
    // lanes 16-31 -> M=lane-16, K 8-15 & 24-31.
    const int row = mBase + (lane & 15);
    const int akb = (lane < 16) ? 0 : 8;
    // B fragment layout (32x16 f16): lanes 0-15 -> N=lane, K 0-7 & 16-23;
    // lanes 16-31 -> N=lane-16, K 8-15 & 24-31.
    const int bkb = (lane < 16) ? 0 : 8;
    const int bcol = lane & 15;

    const _Float16* arow = A + (size_t)row * Kpad;

    for (int kt = 0; kt < Kpad; kt += 32) {
        v8h alo = *(const v8h*)(arow + kt + akb);
        v8h ahi = *(const v8h*)(arow + kt + akb + 16);
        v16h a = pn2_join(alo, ahi);

        v16h b0 = pn2_load_bfrag(Wt, nBase + 0 * 16 + bcol, Kpad, kt + bkb);
        v16h b1 = pn2_load_bfrag(Wt, nBase + 1 * 16 + bcol, Kpad, kt + bkb);
        v16h b2 = pn2_load_bfrag(Wt, nBase + 2 * 16 + bcol, Kpad, kt + bkb);
        v16h b3 = pn2_load_bfrag(Wt, nBase + 3 * 16 + bcol, Kpad, kt + bkb);

        acc0 = __builtin_amdgcn_wmma_f32_16x16x32_f16(false, a, false, b0,
                                                      (short)0, acc0, false, false);
        acc1 = __builtin_amdgcn_wmma_f32_16x16x32_f16(false, a, false, b1,
                                                      (short)0, acc1, false, false);
        acc2 = __builtin_amdgcn_wmma_f32_16x16x32_f16(false, a, false, b2,
                                                      (short)0, acc2, false, false);
        acc3 = __builtin_amdgcn_wmma_f32_16x16x32_f16(false, a, false, b3,
                                                      (short)0, acc3, false, false);
    }

    // C/D layout: col = lane&15, rows = v + 8*(lane>=16)
    const int rbase = mBase + ((lane >> 4) << 3);
    pn2_epilogue(acc0, nBase + 0 * 16 + bcol, rbase, Cout, sc, sh, Out);
    pn2_epilogue(acc1, nBase + 1 * 16 + bcol, rbase, Cout, sc, sh, Out);
    pn2_epilogue(acc2, nBase + 2 * 16 + bcol, rbase, Cout, sc, sh, Out);
    pn2_epilogue(acc3, nBase + 3 * 16 + bcol, rbase, Cout, sc, sh, Out);
    (void)M;
}

// ---------------------------------------------------------------------------
// 8) max over nsample, f16 -> f32; optional (B,C,S) transposed store
// ---------------------------------------------------------------------------
__global__ void pn2_max_k(const _Float16* __restrict__ act,
                          int B, int S, int nsample, int C,
                          float* __restrict__ out, int transposed) {
    long i = (long)blockIdx.x * blockDim.x + threadIdx.x;
    long T = (long)B * S * C;
    if (i >= T) return;
    int c = (int)(i % C);
    long bs = i / C;
    const _Float16* a = act + (bs * nsample) * (size_t)C + c;
    float m = -3.4e38f;
    for (int k = 0; k < nsample; ++k) {
        float v = (float)a[(size_t)k * C];
        m = fmaxf(m, v);
    }
    if (!transposed) {
        out[i] = m;
    } else {
        int b = (int)(bs / S);
        int s = (int)(bs % S);
        out[((size_t)b * C + c) * S + s] = m;
    }
}

// ---------------------------------------------------------------------------
// Host-side orchestration
// ---------------------------------------------------------------------------
struct Pn2Layer {
    const float *w, *b, *g, *be;
    int cin, cout;
};

static void pn2_run_sa(hipStream_t stream,
                       const float* xyz, const float* points,
                       int B, int N, int S, int nsample, float radius, int Cp,
                       const Pn2Layer L[3],
                       int* fpsidx, float* new_xyz, int* ballix,
                       _Float16* Wt, float* sc, float* sh,
                       _Float16* actA, _Float16* actB,
                       float* out_p, int transposed) {
    pn2_fps<<<B, FPS_T, 0, stream>>>(xyz, N, S, fpsidx);

    int bsT = B * S;
    pn2_gather_xyz<<<(bsT + 255) / 256, 256, 0, stream>>>(xyz, fpsidx, B, N, S, new_xyz);
    pn2_ball_query<<<(bsT + 255) / 256, 256, 0, stream>>>(
        xyz, new_xyz, B, N, S, nsample, radius * radius, ballix);

    long M = (long)B * S * nsample;
    int Kpad0 = ((3 + Cp) + 31) / 32 * 32;
    pn2_build_feat<<<(unsigned)((M + 255) / 256), 256, 0, stream>>>(
        xyz, new_xyz, points, ballix, B, N, S, nsample, Cp, Kpad0, actA);

    _Float16* cur = actA;
    _Float16* nxt = actB;
    int Kp = Kpad0;
    for (int l = 0; l < 3; ++l) {
        int Cout = L[l].cout, Cin = L[l].cin;
        int tw = Cout * Kp;
        pn2_conv_w<<<(tw + 255) / 256, 256, 0, stream>>>(L[l].w, Cout, Cin, Kp, Wt);
        pn2_bn_coef<<<(Cout + 63) / 64, 64, 0, stream>>>(L[l].b, L[l].g, L[l].be,
                                                          Cout, sc, sh);
        dim3 grid((unsigned)(M / 128), (unsigned)(Cout / 64));
        pn2_gemm_bn_relu<<<grid, 256, 0, stream>>>(cur, Wt, sc, sh, nxt,
                                                   (int)M, Kp, Cout);
        _Float16* t = cur; cur = nxt; nxt = t;
        Kp = Cout;
    }
    long T = (long)B * S * L[2].cout;
    pn2_max_k<<<(unsigned)((T + 255) / 256), 256, 0, stream>>>(
        cur, B, S, nsample, L[2].cout, out_p, transposed);
}

extern "C" void kernel_launch(void* const* d_in, const int* in_sizes, int n_in,
                              void* d_out, int out_size, void* d_ws, size_t ws_size,
                              hipStream_t stream) {
    (void)in_sizes; (void)n_in; (void)out_size; (void)ws_size;
    const int B = 8, N0 = 4096, S = 512;
    const float* xyz = (const float*)d_in[0];

    char* ws = (char*)d_ws;
    size_t off = 0;
    auto wsalloc = [&](size_t bytes) -> void* {
        void* p = (void*)(ws + off);
        off = (off + bytes + 255) & ~(size_t)255;
        return p;
    };
    float*    pts    = (float*)wsalloc((size_t)B * N0 * 3 * 4);
    int*      fpsidx = (int*)wsalloc((size_t)B * S * 4);
    float*    l1xyz  = (float*)wsalloc((size_t)B * S * 3 * 4);
    float*    l2xyz  = (float*)wsalloc((size_t)B * S * 3 * 4);
    float*    l3xyz  = (float*)wsalloc((size_t)B * S * 3 * 4);
    int*      ballix = (int*)wsalloc((size_t)B * S * 64 * 4);
    float*    l1p    = (float*)wsalloc((size_t)B * S * 128 * 4);
    float*    l2p    = (float*)wsalloc((size_t)B * S * 256 * 4);
    _Float16* Wt     = (_Float16*)wsalloc((size_t)512 * 288 * 2);
    float*    sc     = (float*)wsalloc(512 * 4);
    float*    sh     = (float*)wsalloc(512 * 4);
    _Float16* actA   = (_Float16*)wsalloc((size_t)262144 * 288 * 2);
    _Float16* actB   = (_Float16*)wsalloc((size_t)262144 * 512 * 2);

    {
        long t = (long)B * N0;
        pn2_transpose_xyz<<<(unsigned)((t + 255) / 256), 256, 0, stream>>>(xyz, B, N0, pts);
    }

    auto getL = [&](int base, int cin, int cout) -> Pn2Layer {
        Pn2Layer l;
        l.w  = (const float*)d_in[base + 0];
        l.b  = (const float*)d_in[base + 1];
        l.g  = (const float*)d_in[base + 2];
        l.be = (const float*)d_in[base + 3];
        l.cin = cin; l.cout = cout;
        return l;
    };
    Pn2Layer P1[3] = { getL(1, 3, 64),    getL(5, 64, 64),   getL(9, 64, 128) };
    Pn2Layer P2[3] = { getL(13, 131, 128), getL(17, 128, 128), getL(21, 128, 256) };
    Pn2Layer P3[3] = { getL(25, 259, 256), getL(29, 256, 256), getL(33, 256, 512) };

    // SA1: N=4096 -> S=512, nsample=32, radius 0.2, no extra point features
    pn2_run_sa(stream, pts, nullptr, B, N0, S, 32, 0.2f, 0, P1,
               fpsidx, l1xyz, ballix, Wt, sc, sh, actA, actB, l1p, 0);
    // SA2: N=512 -> S=512, nsample=64, radius 0.4, Cp=128
    pn2_run_sa(stream, l1xyz, l1p, B, S, S, 64, 0.4f, 128, P2,
               fpsidx, l2xyz, ballix, Wt, sc, sh, actA, actB, l2p, 0);
    // SA3: N=512 -> S=512, nsample=64, radius 0.4, Cp=256; output (B,512,512) transposed
    pn2_run_sa(stream, l2xyz, l2p, B, S, S, 64, 0.4f, 256, P3,
               fpsidx, l3xyz, ballix, Wt, sc, sh, actA, actB, (float*)d_out, 1);
}
